// HGNN_64587718197893
// MI455X (gfx1250) — matrix operand order repaired and checked by reading, not compiled, not run on hardware
//
#include <hip/hip_runtime.h>
#include <cstdint>
#include <cstddef>

#define NB 100000
#define NS 50000
#define DIM 128
#define NHEADS 8
#define E_BB 800000
#define E_SB 400000
#define E_BS 400000
#define NLAYERS 2

typedef __attribute__((ext_vector_type(16))) __bf16 v16bf;
typedef __attribute__((ext_vector_type(8)))  float  v8f;
typedef __attribute__((ext_vector_type(16))) unsigned short us16;

__device__ __forceinline__ unsigned short f2bf(float f) {
  unsigned int u = __float_as_uint(f);
  if ((u & 0x7F800000u) == 0x7F800000u) return (unsigned short)(u >> 16); // Inf/NaN: truncate
  unsigned int r = u + 0x7FFFu + ((u >> 16) & 1u);                        // round-nearest-even
  return (unsigned short)(r >> 16);
}

// ---------------------------------------------------------------------------
// Weight prep: build fused, bf16, WMMA-B-fragment-swizzled weight blocks.
// B-fragment layout per (tile t, kchunk kc): 32 lanes x 16 bf16 contiguous;
// lane l -> N = t*16 + (l%16), element e -> K = kc*32 + 16*(l/16) + e.
// b-dst fused K=384: [Wl_bb ; Wl_sb ; Wr_bb+Wr_sb]; s-dst K=256: [Wl_bs ; Wr_bs]
// ---------------------------------------------------------------------------
#define NSWB (NLAYERS*8*12*512)
#define NSWS (NLAYERS*8*8*512)

__global__ void prep_weights(const float* __restrict__ Wl, const float* __restrict__ bl,
                             const float* __restrict__ Wr,
                             unsigned short* __restrict__ BswB, unsigned short* __restrict__ BswS,
                             float* __restrict__ biasB, float* __restrict__ biasS) {
  int id = blockIdx.x * blockDim.x + threadIdx.x;
  if (id < NSWB) {
    int e = id & 15, lane = (id >> 4) & 31, rem = id >> 9;
    int kc = rem % 12; rem /= 12; int t = rem & 7; int l = rem >> 3;
    int K = kc * 32 + ((lane >> 4) << 4) + e;
    int n = t * 16 + (lane & 15);
    float w;
    if (K < 128)      w = Wl[((l*3 + 0)*128 + K) * 128 + n];
    else if (K < 256) w = Wl[((l*3 + 1)*128 + (K-128)) * 128 + n];
    else              w = Wr[((l*3 + 0)*128 + (K-256)) * 128 + n]
                        + Wr[((l*3 + 1)*128 + (K-256)) * 128 + n];
    BswB[id] = f2bf(w);
  } else if (id < NSWB + NSWS) {
    int j = id - NSWB;
    int e = j & 15, lane = (j >> 4) & 31, rem = j >> 9;
    int kc = rem & 7; rem >>= 3; int t = rem & 7; int l = rem >> 3;
    int K = kc * 32 + ((lane >> 4) << 4) + e;
    int n = t * 16 + (lane & 15);
    float w = (K < 128) ? Wl[((l*3 + 2)*128 + K) * 128 + n]
                        : Wr[((l*3 + 2)*128 + (K-128)) * 128 + n];
    BswS[j] = f2bf(w);
  } else if (id < NSWB + NSWS + NLAYERS * 128) {
    int j = id - NSWB - NSWS; int l = j >> 7, n = j & 127;
    biasB[j] = bl[(l*3 + 0)*128 + n] + bl[(l*3 + 1)*128 + n];
  } else if (id < NSWB + NSWS + 2 * NLAYERS * 128) {
    int j = id - NSWB - NSWS - NLAYERS*128; int l = j >> 7, n = j & 127;
    biasS[j] = bl[(l*3 + 2)*128 + n];
  }
}

__global__ void zero_kernel(float* __restrict__ p, long n) {
  long i = blockIdx.x * (long)blockDim.x + threadIdx.x;
  long stride = (long)gridDim.x * blockDim.x;
  for (; i < n; i += stride) p[i] = 0.0f;
}

// ---------------------------------------------------------------------------
// Scatter-sum + count: one wave32 per edge; lane carries a float4 (16B) slice.
// ---------------------------------------------------------------------------
__global__ void scatter_accum(const float* __restrict__ xsrc, const int* __restrict__ ei,
                              int E, float* __restrict__ acc, float* __restrict__ cnt) {
  int gid = blockIdx.x * blockDim.x + threadIdx.x;
  int edge = gid >> 5, lane = gid & 31;
  if (edge >= E) return;
  int src = ei[edge];
  int dst = ei[E + edge];
  float4 v = ((const float4*)(xsrc + (size_t)src * DIM))[lane];
  float* a = acc + (size_t)dst * DIM + lane * 4;
  atomicAdd(a + 0, v.x); atomicAdd(a + 1, v.y);
  atomicAdd(a + 2, v.z); atomicAdd(a + 3, v.w);
  if (lane == 0) atomicAdd(cnt + dst, 1.0f);
}

// ---------------------------------------------------------------------------
// Fused SAGE GEMM via v_wmma_f32_16x16x32_bf16.
// Block = 256 threads (8 waves), 16 nodes x 128 outputs. Wave w -> N-tile w.
// A tile staged in LDS pre-swizzled to the same contiguous-k fragment order as B:
//   index (kc*32 + lane)*16 + e  holds  A[lane%16][kc*32 + 16*(lane/16) + e]
// so each lane reads one contiguous 32B us16 per K-chunk.
// ---------------------------------------------------------------------------
template <int KA, int KC, bool HAS2>
__launch_bounds__(256)
__global__ void sage_gemm(const float* __restrict__ acc0, const float* __restrict__ cnt0,
                          const float* __restrict__ acc1, const float* __restrict__ cnt1,
                          const float* __restrict__ xdst,
                          const unsigned short* __restrict__ Bsw,
                          const float* __restrict__ bias,
                          float* __restrict__ out) {
  __shared__ __align__(32) unsigned short sA[16 * KA];
  __shared__ float sInv[32];
  __shared__ float sBias[128];
  const int tid = threadIdx.x;
  const int rb  = blockIdx.x;

  if (tid < 16) {
    sInv[tid] = 1.0f / fmaxf(cnt0[rb * 16 + tid], 1.0f);
  } else if (HAS2 && tid < 32) {
    sInv[tid] = 1.0f / fmaxf(cnt1[rb * 16 + (tid - 16)], 1.0f);
  }
  if (tid >= 128) sBias[tid - 128] = bias[tid - 128];
  __syncthreads();

  // Stage A (scatter-mean division folded in), directly in fragment order.
  for (int d = tid; d < 16 * KA; d += 256) {
    int e = d & 15, lane = (d >> 4) & 31, kc = d >> 9;
    int m = lane & 15;
    int K = kc * 32 + 16 * (lane >> 4) + e;
    int node = rb * 16 + m;
    float v;
    if (K < 128)                 v = acc0[(size_t)node * DIM + K] * sInv[m];
    else if (HAS2 && K < 256)    v = acc1[(size_t)node * DIM + (K - 128)] * sInv[16 + m];
    else                         v = xdst[(size_t)node * DIM + (K - (HAS2 ? 256 : 128))];
    sA[d] = f2bf(v);
  }
  __syncthreads();

  const int wave = tid >> 5, lane = tid & 31;
  v8f c = {0.f, 0.f, 0.f, 0.f, 0.f, 0.f, 0.f, 0.f};
  const unsigned short* bp = Bsw + (size_t)wave * KC * 512 + lane * 16;
#pragma unroll
  for (int kc = 0; kc < KC; ++kc) {
    us16 av = *(const us16*)(&sA[(kc * 32 + lane) * 16]);
    us16 bv = *(const us16*)(bp + kc * 512);
    v16bf a = __builtin_bit_cast(v16bf, av);
    v16bf b = __builtin_bit_cast(v16bf, bv);
    c = __builtin_amdgcn_wmma_f32_16x16x32_bf16(false, a, false, b, (short)0, c, false, false);
  }

  // C/D layout: lane 0-15 -> M=r, lane 16-31 -> M=r+8; N = lane%16.
  const int n = wave * 16 + (lane & 15);
  const int mbase = 8 * (lane >> 4);
  const float bsv = sBias[n];
#pragma unroll
  for (int r = 0; r < 8; ++r) {
    int node = rb * 16 + mbase + r;
    float v = c[r] + bsv;
    v = v > 0.f ? v : 0.01f * v;  // leaky_relu(0.01)
    out[(size_t)node * DIM + n] = v;
  }
}

// ---------------------------------------------------------------------------
// Readout: out[node,h] = dot(x_b[node], Wh[h]) + bh[h]   ([100000,128]x[128,8])
// ---------------------------------------------------------------------------
__global__ void head_kernel(const float* __restrict__ xb, const float* __restrict__ Wh,
                            const float* __restrict__ bh, float* __restrict__ out) {
  int i = blockIdx.x * blockDim.x + threadIdx.x;
  if (i >= NB * NHEADS) return;
  int node = i >> 3, h = i & 7;
  const float* xr = xb + (size_t)node * DIM;
  const float* wr = Wh + h * DIM;
  float s = bh[h];
#pragma unroll 8
  for (int k = 0; k < DIM; ++k) s += xr[k] * wr[k];
  out[i] = s;
}

// ---------------------------------------------------------------------------
extern "C" void kernel_launch(void* const* d_in, const int* in_sizes, int n_in,
                              void* d_out, int out_size, void* d_ws, size_t ws_size,
                              hipStream_t stream) {
  const float* x_b = (const float*)d_in[0];
  const float* x_s = (const float*)d_in[1];
  const float* Wl  = (const float*)d_in[2];
  const float* bl  = (const float*)d_in[3];
  const float* Wr  = (const float*)d_in[4];
  const float* Wh  = (const float*)d_in[5];
  const float* bh  = (const float*)d_in[6];
  const int* ei_bb = (const int*)d_in[7];
  const int* ei_sb = (const int*)d_in[8];
  const int* ei_bs = (const int*)d_in[9];
  (void)in_sizes; (void)n_in; (void)out_size; (void)ws_size;

  uint8_t* base = (uint8_t*)d_ws;
  size_t off = 0;
  auto take = [&](size_t bytes) -> void* {
    void* p = base + off;
    off = (off + bytes + 255) & ~(size_t)255;
    return p;
  };

  unsigned short* BswB = (unsigned short*)take((size_t)NSWB * 2);
  unsigned short* BswS = (unsigned short*)take((size_t)NSWS * 2);
  float* biasB = (float*)take((size_t)NLAYERS * 128 * 4);
  float* biasS = (float*)take((size_t)NLAYERS * 128 * 4);

  // Contiguous accumulator region (zeroed once per layer with a single kernel)
  const size_t accFloats = (size_t)NB*DIM + NB + (size_t)NB*DIM + NB + (size_t)NS*DIM + NS;
  float* accBB = (float*)take(accFloats * 4);
  float* cntBB = accBB + (size_t)NB * DIM;
  float* accSB = cntBB + NB;
  float* cntSB = accSB + (size_t)NB * DIM;
  float* accBS = cntSB + NB;
  float* cntBS = accBS + (size_t)NS * DIM;

  float* xbA = (float*)take((size_t)NB * DIM * 4);
  float* xbB = (float*)take((size_t)NB * DIM * 4);
  float* xsA = (float*)take((size_t)NS * DIM * 4);
  float* xsB = (float*)take((size_t)NS * DIM * 4);

  {
    int total = NSWB + NSWS + 2 * NLAYERS * 128;
    prep_weights<<<(total + 255) / 256, 256, 0, stream>>>(Wl, bl, Wr, BswB, BswS, biasB, biasS);
  }

  const float* xbin = x_b;
  const float* xsin = x_s;
  float* xbout[NLAYERS] = {xbA, xbB};
  float* xsout[NLAYERS] = {xsA, xsB};

  for (int l = 0; l < NLAYERS; ++l) {
    zero_kernel<<<2048, 256, 0, stream>>>(accBB, (long)accFloats);
    scatter_accum<<<(E_BB * 32 + 255) / 256, 256, 0, stream>>>(xbin, ei_bb, E_BB, accBB, cntBB);
    scatter_accum<<<(E_SB * 32 + 255) / 256, 256, 0, stream>>>(xsin, ei_sb, E_SB, accSB, cntSB);
    scatter_accum<<<(E_BS * 32 + 255) / 256, 256, 0, stream>>>(xbin, ei_bs, E_BS, accBS, cntBS);

    sage_gemm<384, 12, true><<<NB / 16, 256, 0, stream>>>(
        accBB, cntBB, accSB, cntSB, xbin,
        BswB + (size_t)l * 8 * 12 * 512, biasB + l * 128, xbout[l]);
    sage_gemm<256, 8, false><<<NS / 16, 256, 0, stream>>>(
        accBS, cntBS, nullptr, nullptr, xsin,
        BswS + (size_t)l * 8 * 8 * 512, biasS + l * 128, xsout[l]);

    xbin = xbout[l];
    xsin = xsout[l];
  }

  head_kernel<<<(NB * NHEADS + 255) / 256, 256, 0, stream>>>(xbin, Wh, bh, (float*)d_out);
}
